// GLALOB_84748294684696
// MI455X (gfx1250) — matrix operand order repaired and checked by reading, not compiled, not run on hardware
//
#include <hip/hip_runtime.h>

// MI455X / gfx1250 implementation of the GLA forward pass.
// Heavy lifting: bf16 WMMA GEMM (v_wmma_f32_16x16x32_bf16), f32 accumulate.
// Fast-path tile staging uses GLOBAL_LOAD_ASYNC_TO_LDS_B128 (ASYNCcnt),
// double-buffered so the async DMA of tile k+1 overlaps WMMA on tile k.
// Everything runs on `stream`; scratch comes from d_ws via a bump allocator.

typedef unsigned short u16;
typedef __attribute__((ext_vector_type(16))) __bf16 v16bf;
typedef __attribute__((ext_vector_type(8)))  float  v8f;
typedef __attribute__((ext_vector_type(4)))  unsigned int v4u;
typedef __attribute__((ext_vector_type(4)))  int v4i;

#define EPSF 1e-5f
#define AS1 __attribute__((address_space(1)))
#define AS3 __attribute__((address_space(3)))

#if __has_builtin(__builtin_amdgcn_global_load_async_to_lds_b128) && \
    __has_builtin(__builtin_amdgcn_s_wait_asynccnt)
#define USE_ASYNC_COPY 1
#else
#define USE_ASYNC_COPY 0
#endif

static __device__ __forceinline__ u16 f2b(float f) {
  unsigned int u = __float_as_uint(f);
  u += 0x7fffu + ((u >> 16) & 1u);          // round-to-nearest-even bf16
  return (u16)(u >> 16);
}
static __device__ __forceinline__ float geluf(float v) {
  return 0.5f * v * (1.0f + erff(v * 0.70710678118654752f));
}

union Frag { v16bf v; v4u u[2]; };

// ---------------------------------------------------------------------------
// Tiled bf16 WMMA GEMM:  C = act(scale * (A @ W^T) + bias + addend)
// A : (M,K) bf16 row-major.  Wt : (N,K) bf16 row-major (W transposed).
// Cf: optional (M,N) f32 out.  Cb: optional (M,N) bf16 out.
// act: 0 none, 1 exact GELU, 2 log_sigmoid(x)/16.
// Block: 256 threads = 8 waves. Tile 128(M)x128(N), K-step 32, double-buffered.
// Wave (wm,wn) owns a 32x64 sub-tile -> 2x4 wmma 16x16x32 per K-step.
// M is always a multiple of 128 by launch contract.
// ---------------------------------------------------------------------------
__global__ __launch_bounds__(256) void gemm_bf16_wmma(
    const u16* __restrict__ A, const u16* __restrict__ Wt,
    const float* __restrict__ bias, const float* __restrict__ addend,
    float* __restrict__ Cf, u16* __restrict__ Cb,
    int M, int K, int N, int act, float scale)
{
  __shared__ u16 ldsA[2][128 * 32];
  __shared__ u16 ldsB[2][128 * 32];
  const int tid   = threadIdx.x;
  const int mBase = blockIdx.x * 128;
  const int nBase = blockIdx.y * 128;
  const int wave  = tid >> 5, lane = tid & 31;
  const int wm    = wave >> 1, wn = wave & 1;
  const int half  = lane >> 4, r = lane & 15;

  // Block-uniform fast-path condition: full 128-row N tile, K multiple of 32.
  const bool fast = ((K & 31) == 0) && (nBase + 128 <= N);

  // Stage one 128x32 K-tile of A and W^T into LDS buffer `buf`.
  auto stage = [&](int k0, int buf) {
    if (fast) {
#if USE_ASYNC_COPY
      // Async DMA to LDS (ASYNCcnt); each thread moves 16B of A and 16B of B
      // per pass. No wait here -- the copy overlaps WMMA on the other buffer.
      #pragma unroll
      for (int p = 0; p < 2; ++p) {
        int e8  = p * 256 + tid;
        int row = e8 >> 2;
        int col = (e8 & 3) * 8;
        __builtin_amdgcn_global_load_async_to_lds_b128(
            (AS1 v4i*)&A[(size_t)(mBase + row) * K + k0 + col],
            (AS3 v4i*)&ldsA[buf][row * 32 + col], 0, 0);
        __builtin_amdgcn_global_load_async_to_lds_b128(
            (AS1 v4i*)&Wt[(size_t)(nBase + row) * K + k0 + col],
            (AS3 v4i*)&ldsB[buf][row * 32 + col], 0, 0);
      }
#else
      #pragma unroll
      for (int p = 0; p < 2; ++p) {
        int e8  = p * 256 + tid;
        int row = e8 >> 2;
        int col = (e8 & 3) * 8;
        *(v4u*)&ldsA[buf][row * 32 + col] =
            *(const v4u*)&A[(size_t)(mBase + row) * K + k0 + col];
        *(v4u*)&ldsB[buf][row * 32 + col] =
            *(const v4u*)&Wt[(size_t)(nBase + row) * K + k0 + col];
      }
#endif
    } else {
      // Ragged edge path (K=16 low-rank, N in {3,16,64} tails): guarded scalars.
      #pragma unroll
      for (int p = 0; p < 2; ++p) {
        int e8  = p * 256 + tid;
        int row = e8 >> 2;
        int col = (e8 & 3) * 8;
        #pragma unroll
        for (int j = 0; j < 8; ++j) {
          int kk = k0 + col + j;
          ldsA[buf][row * 32 + col + j] =
              (kk < K) ? A[(size_t)(mBase + row) * K + kk] : (u16)0;
          int gn = nBase + row;
          ldsB[buf][row * 32 + col + j] =
              (gn < N && kk < K) ? Wt[(size_t)gn * K + kk] : (u16)0;
        }
      }
    }
  };

  v8f acc[2][4];
  #pragma unroll
  for (int i = 0; i < 2; ++i)
    #pragma unroll
    for (int j = 0; j < 4; ++j)
      #pragma unroll
      for (int e = 0; e < 8; ++e) acc[i][j][e] = 0.0f;

  // Prologue: stage tile 0, make it visible to the whole workgroup.
  stage(0, 0);
#if USE_ASYNC_COPY
  if (fast) __builtin_amdgcn_s_wait_asynccnt(0);
#endif
  __syncthreads();

  int cur = 0;
  for (int k0 = 0; k0 < K; k0 += 32) {
    const bool have_next = (k0 + 32 < K);
    if (k0 + 64 < K) {
      // two tiles ahead: async covers k+1, prefetch warms k+2 (global_prefetch_b8)
      __builtin_prefetch(&A[(size_t)(mBase + (tid >> 1)) * (size_t)K + k0 + 64], 0, 3);
    }
    if (have_next) stage(k0 + 32, cur ^ 1);   // overlaps with WMMA below

    // A frag: lane r holds row (base + r); lo lanes K{0..7,16..23}, hi K{8..15,24..31}
    Frag fa[2], fb[4];
    #pragma unroll
    for (int i = 0; i < 2; ++i) {
      int base = (wm * 32 + i * 16 + r) * 32 + half * 8;
      fa[i].u[0] = *(const v4u*)&ldsA[cur][base];
      fa[i].u[1] = *(const v4u*)&ldsA[cur][base + 16];
    }
    // B frag: lane r holds output-col (base + r); lo lanes K{0..7,16..23}, hi K{8..15,24..31}
    #pragma unroll
    for (int j = 0; j < 4; ++j) {
      int base = (wn * 64 + j * 16 + r) * 32 + half * 8;
      fb[j].u[0] = *(const v4u*)&ldsB[cur][base];
      fb[j].u[1] = *(const v4u*)&ldsB[cur][base + 16];
    }
    #pragma unroll
    for (int i = 0; i < 2; ++i)
      #pragma unroll
      for (int j = 0; j < 4; ++j)
        acc[i][j] = __builtin_amdgcn_wmma_f32_16x16x32_bf16(
            false, fa[i].v, false, fb[j].v, (short)0, acc[i][j], false, false);

#if USE_ASYNC_COPY
    if (fast && have_next) __builtin_amdgcn_s_wait_asynccnt(0);
#endif
    __syncthreads();
    cur ^= 1;
  }

  // Epilogue. C/D layout: element e of v8f in lane l -> M = e + 8*(l>>4), N = l&15.
  #pragma unroll
  for (int i = 0; i < 2; ++i) {
    #pragma unroll
    for (int j = 0; j < 4; ++j) {
      int col = nBase + wn * 64 + j * 16 + r;
      if (col >= N) continue;
      float bv = bias ? bias[col] : 0.0f;
      #pragma unroll
      for (int e = 0; e < 8; ++e) {
        int row = mBase + wm * 32 + i * 16 + half * 8 + e;
        float vv = acc[i][j][e] * scale + bv;
        if (addend) vv += addend[(size_t)row * N + col];
        if (act == 1) vv = geluf(vv);
        else if (act == 2) {
          float ls = fminf(vv, 0.0f) - log1pf(expf(-fabsf(vv)));
          vv = ls * (1.0f / 16.0f);
        }
        if (Cf) Cf[(size_t)row * N + col] = vv;
        if (Cb) Cb[(size_t)row * N + col] = f2b(vv);
      }
    }
  }
}

// ---------------------------------------------------------------------------
// Weight transpose + f32->bf16:  wt[n*K+k] = bf16(w[k*N+n])
// ---------------------------------------------------------------------------
__global__ void wconv_kernel(const float* __restrict__ w, u16* __restrict__ wt,
                             int K, int N)
{
  int e = blockIdx.x * 256 + threadIdx.x;
  if (e >= K * N) return;
  int k = e / N, n = e - k * N;
  wt[(size_t)n * K + k] = f2b(w[e]);
}

// ---------------------------------------------------------------------------
// BiN: input x (B,S,F). Per-(b,f) stats over s and per-(b,s) stats over f,
// sample std (ddof=1), clamp std<1e-4 -> 1. Output (B,S,F).
// One block per b; S=256 threads.
// ---------------------------------------------------------------------------
__global__ __launch_bounds__(256) void bin_norm_kernel(
    const float* __restrict__ x,
    const float* __restrict__ B1, const float* __restrict__ l1,
    const float* __restrict__ B2, const float* __restrict__ l2,
    const float* __restrict__ y1, const float* __restrict__ y2,
    float* __restrict__ out)
{
  const int S = 256, F = 40;
  int b = blockIdx.x, tid = threadIdx.x;
  __shared__ float xs[256 * 40];
  __shared__ float mS[256], sS[256];
  __shared__ float mF[40], sF[40];
  const float* xb = x + (size_t)b * S * F;
  for (int e = tid; e < S * F; e += 256) xs[e] = xb[e];
  __syncthreads();
  {  // per-s over f (n=40)
    int s = tid;
    float sum = 0.f, sum2 = 0.f;
    for (int f = 0; f < F; ++f) { float v = xs[s * F + f]; sum += v; sum2 += v * v; }
    float m = sum / 40.0f;
    float var = (sum2 - 40.0f * m * m) / 39.0f;
    float sd = sqrtf(fmaxf(var, 0.0f));
    if (sd < 1e-4f) sd = 1.0f;
    mS[s] = m; sS[s] = sd;
  }
  if (tid < F) {  // per-f over s (n=256)
    int f = tid;
    float sum = 0.f, sum2 = 0.f;
    for (int s = 0; s < S; ++s) { float v = xs[s * F + f]; sum += v; sum2 += v * v; }
    float m = sum / 256.0f;
    float var = (sum2 - 256.0f * m * m) / 255.0f;
    float sd = sqrtf(fmaxf(var, 0.0f));
    if (sd < 1e-4f) sd = 1.0f;
    mF[f] = m; sF[f] = sd;
  }
  __syncthreads();
  float Y1 = y1[0], Y2 = y2[0];
  for (int e = tid; e < S * F; e += 256) {
    int s = e / F, f = e - s * F;
    float v = xs[e];
    float X2 = l2[f] * ((v - mF[f]) / sF[f]) + B2[f];
    float X1 = l1[s] * ((v - mS[s]) / sS[s]) + B1[s];
    out[(size_t)b * S * F + e] = Y1 * X1 + Y2 * X2;
  }
}

// ---------------------------------------------------------------------------
// x0 = y @ emb_w + emb_b + sin_pos_emb. One block per (b,s), thread = d.
// ---------------------------------------------------------------------------
__global__ __launch_bounds__(256) void embed_kernel(
    const float* __restrict__ y, const float* __restrict__ emb_w,
    const float* __restrict__ emb_b, float* __restrict__ xf, u16* __restrict__ xb)
{
  int bs = blockIdx.x;
  int s = bs & 255;
  int d = threadIdx.x;
  __shared__ float yr[40];
  if (d < 40) yr[d] = y[(size_t)bs * 40 + d];
  __syncthreads();
  float acc = emb_b[d];
  #pragma unroll 8
  for (int f = 0; f < 40; ++f) acc += yr[f] * emb_w[f * 256 + d];
  int i = d >> 1;
  float denom = powf(10000.0f, (2.0f * (float)i) / 256.0f);
  float ang = (float)s / denom;
  acc += (d & 1) ? cosf(ang) : sinf(ang);
  xf[(size_t)bs * 256 + d] = acc;
  xb[(size_t)bs * 256 + d] = f2b(acc);
}

// ---------------------------------------------------------------------------
// Gated linear attention recurrence. One block per (b,h); 256 threads.
// Thread t owns v-column (t&63) and 8 k-rows (t>>6)*8.. . State in registers.
// ---------------------------------------------------------------------------
__global__ __launch_bounds__(256) void scan_kernel(
    const float* __restrict__ q, const float* __restrict__ k,
    const float* __restrict__ v, const float* __restrict__ gk,
    float* __restrict__ o, int T)
{
  int b = blockIdx.x >> 2;
  int h = blockIdx.x & 3;
  int tid = threadIdx.x;
  int vi = tid & 63;
  int kg = tid >> 6;
  float st[8];
  #pragma unroll
  for (int j = 0; j < 8; ++j) st[j] = 0.0f;
  __shared__ float sq[32], sk[32], sg[32], sv[64], sp[256];
  for (int t = 0; t < T; ++t) {
    size_t rk = ((size_t)(b * T + t)) * 128 + h * 32;
    size_t rv = ((size_t)(b * T + t)) * 256 + h * 64;
    if (tid < 32) { sq[tid] = q[rk + tid]; sk[tid] = k[rk + tid]; sg[tid] = gk[rk + tid]; }
    else if (tid < 96) { sv[tid - 32] = v[rv + (tid - 32)]; }
    __syncthreads();
    float vt = sv[vi];
    float accv = 0.0f;
    #pragma unroll
    for (int j = 0; j < 8; ++j) {
      int kk = kg * 8 + j;
      st[j] = expf(sg[kk]) * st[j] + sk[kk] * vt;
      accv += sq[kk] * st[j];
    }
    sp[kg * 64 + vi] = accv;
    __syncthreads();
    if (kg == 0) o[rv + vi] = sp[vi] + sp[64 + vi] + sp[128 + vi] + sp[192 + vi];
  }
}

// ---------------------------------------------------------------------------
// GroupNorm(RMS over 64-ch head) * gn_w, then * g * sigmoid(g) -> bf16.
// ---------------------------------------------------------------------------
__global__ __launch_bounds__(256) void gate_norm_kernel(
    const float* __restrict__ o, const float* __restrict__ g,
    const float* __restrict__ gnw, u16* __restrict__ out)
{
  int row = blockIdx.x, c = threadIdx.x, h = c >> 6;
  __shared__ float red[256];
  float ov = o[(size_t)row * 256 + c];
  float gv = g[(size_t)row * 256 + c];
  red[c] = ov * ov;
  __syncthreads();
  for (int s2 = 32; s2 > 0; s2 >>= 1) {
    if ((c & 63) < s2) red[c] += red[c + s2];
    __syncthreads();
  }
  float ms = red[h << 6] * (1.0f / 64.0f);
  float on = ov * rsqrtf(ms + EPSF) * gnw[c & 63];
  float sg = 1.0f / (1.0f + expf(-gv));
  out[(size_t)row * 256 + c] = f2b(on * gv * sg);
}

// ---------------------------------------------------------------------------
// LayerNorm over 256 channels -> bf16.
// ---------------------------------------------------------------------------
__global__ __launch_bounds__(256) void layernorm_kernel(
    const float* __restrict__ x, const float* __restrict__ w,
    const float* __restrict__ b, u16* __restrict__ out)
{
  int row = blockIdx.x, tid = threadIdx.x;
  __shared__ float red[256];
  float v = x[(size_t)row * 256 + tid];
  red[tid] = v;
  __syncthreads();
  for (int s2 = 128; s2 > 0; s2 >>= 1) { if (tid < s2) red[tid] += red[tid + s2]; __syncthreads(); }
  float mu = red[0] * (1.0f / 256.0f);
  __syncthreads();
  float d = v - mu;
  red[tid] = d * d;
  __syncthreads();
  for (int s2 = 128; s2 > 0; s2 >>= 1) { if (tid < s2) red[tid] += red[tid + s2]; __syncthreads(); }
  float var = red[0] * (1.0f / 256.0f);
  float o = d * rsqrtf(var + EPSF) * w[tid] + b[tid];
  out[(size_t)row * 256 + tid] = f2b(o);
}

// ---------------------------------------------------------------------------
// (B,T,C) -> (B,C,T), f32 + bf16 out. T,C multiples of 32. Block 32x8.
// ---------------------------------------------------------------------------
__global__ void transpose_kernel(const float* __restrict__ in,
                                 float* __restrict__ outF, u16* __restrict__ outB,
                                 int T, int C)
{
  __shared__ float tile[32][33];
  int b = blockIdx.z;
  int c0 = blockIdx.x * 32, t0 = blockIdx.y * 32;
  int tx = threadIdx.x, ty = threadIdx.y;
  #pragma unroll
  for (int j = 0; j < 4; ++j) {
    int t = t0 + ty + j * 8;
    tile[ty + j * 8][tx] = in[((size_t)b * T + t) * C + c0 + tx];
  }
  __syncthreads();
  #pragma unroll
  for (int j = 0; j < 4; ++j) {
    int c = c0 + ty + j * 8;
    float v = tile[tx][ty + j * 8];
    size_t o = ((size_t)b * C + c) * T + t0 + tx;
    outF[o] = v;
    outB[o] = f2b(v);
  }
}

// ---------------------------------------------------------------------------
// Host orchestration.
// Input flattening assumption (insertion order of setup_inputs):
//   x, bin{B1,l1,B2,l2,y1,y2}, emb_w, emb_b,
//   8 layers x {Wq,Wk,Wv,Wgk1,Wgk2,bgk2,Wg,gn_w,Wo,w0_w,w0_b,ln_w,ln_b,W1,b1,W2,b2},
//   finals 4 x (W, b).
// ---------------------------------------------------------------------------
extern "C" void kernel_launch(void* const* d_in, const int* in_sizes, int n_in,
                              void* d_out, int out_size, void* d_ws, size_t ws_size,
                              hipStream_t stream)
{
  (void)in_sizes; (void)n_in; (void)out_size; (void)ws_size;
  int idx = 0;
  auto nx = [&]() -> const float* { return (const float*)d_in[idx++]; };

  const float* X  = nx();
  const float* B1 = nx(); const float* l1 = nx();
  const float* B2 = nx(); const float* l2 = nx();
  const float* y1 = nx(); const float* y2 = nx();
  const float* emb_w = nx(); const float* emb_b = nx();
  const float* Lw[8][17];
  for (int i = 0; i < 8; ++i)
    for (int j = 0; j < 17; ++j) Lw[i][j] = nx();
  const float* FW[4]; const float* Fbias[4];
  for (int i = 0; i < 4; ++i) { FW[i] = nx(); Fbias[i] = nx(); }

  char* ws = (char*)d_ws;
  size_t off = 0;
  auto alloc = [&](size_t bytes) -> void* {
    void* p = ws + off;
    off += (bytes + 255) & ~(size_t)255;
    return p;
  };

  const int Bn = 256;
  const size_t Mfull = (size_t)Bn * 256;

  float* Y   = (float*)alloc(Mfull * 40 * 4);
  float* Af  = (float*)alloc(Mfull * 256 * 4);   // current layer input, f32
  u16*   Ab  = (u16*)  alloc(Mfull * 256 * 2);   // current layer input, bf16
  float* Qf  = (float*)alloc(Mfull * 128 * 4);
  float* Kf  = (float*)alloc(Mfull * 128 * 4);
  float* GKf = (float*)alloc(Mfull * 128 * 4);
  float* Vf  = (float*)alloc(Mfull * 256 * 4);
  float* Gf  = (float*)alloc(Mfull * 256 * 4);
  float* Of  = (float*)alloc(Mfull * 256 * 4);
  u16*   OB  = (u16*)  alloc(Mfull * 256 * 2);
  u16*   T1  = (u16*)  alloc(Mfull * 16 * 2);
  u16*   ABf = (u16*)  alloc(Mfull * 256 * 2);
  float* X1  = (float*)alloc(Mfull * 256 * 4);
  u16*   XN  = (u16*)  alloc(Mfull * 256 * 2);
  u16*   H1  = (u16*)  alloc(Mfull * 1024 * 2);
  float* Hf  = (float*)alloc(Mfull * 256 * 4);
  u16*   Hb  = (u16*)  alloc((size_t)Bn * 4096 * 2);
  u16*   Wpool = (u16*)alloc((size_t)6 * 1024 * 1024 * 2);
  u16*   FA  = (u16*)  alloc((size_t)256 * 1024 * 2);
  u16*   FBx = (u16*)  alloc((size_t)256 * 256 * 2);
  u16*   FC  = (u16*)  alloc((size_t)256 * 64 * 2);

  auto gemm = [&](const u16* A, const u16* W, const float* bias, const float* add,
                  float* Cf, u16* Cb, int M, int K, int N, int act, float scale) {
    dim3 g((unsigned)((M + 127) / 128), (unsigned)((N + 127) / 128));
    gemm_bf16_wmma<<<g, 256, 0, stream>>>(A, W, bias, add, Cf, Cb, M, K, N, act, scale);
  };
  auto wconv = [&](const float* w, u16* wt, int K, int N) {
    int n = K * N;
    wconv_kernel<<<(n + 255) / 256, 256, 0, stream>>>(w, wt, K, N);
  };

  // 1) BiN + embedding + positional encoding
  bin_norm_kernel<<<Bn, 256, 0, stream>>>(X, B1, l1, B2, l2, y1, y2, Y);
  embed_kernel<<<Bn * 256, 256, 0, stream>>>(Y, emb_w, emb_b, Af, Ab);

  // 2) 8 GLA sub-layers (feature dim is always 256; T shrinks to 64 after layer 6)
  int curT = 256;
  for (int li = 0; li < 8; ++li) {
    const int Nout = (li >= 6) ? 64 : 256;
    const int M = Bn * curT;
    size_t wo = 0;
    auto wsub = [&](int K, int N) -> u16* { u16* p = Wpool + wo; wo += (size_t)K * N; return p; };
    u16* WqT  = wsub(256, 128);   wconv(Lw[li][0],  WqT, 256, 128);
    u16* WkT  = wsub(256, 128);   wconv(Lw[li][1],  WkT, 256, 128);
    u16* WvT  = wsub(256, 256);   wconv(Lw[li][2],  WvT, 256, 256);
    u16* Wg1T = wsub(256, 16);    wconv(Lw[li][3],  Wg1T, 256, 16);
    u16* Wg2T = wsub(16, 128);    wconv(Lw[li][4],  Wg2T, 16, 128);
    u16* WgT  = wsub(256, 256);   wconv(Lw[li][6],  WgT, 256, 256);
    u16* WoT  = wsub(256, 256);   wconv(Lw[li][8],  WoT, 256, 256);
    u16* W0T  = wsub(256, 256);   wconv(Lw[li][9],  W0T, 256, 256);
    u16* W1T  = wsub(256, 1024);  wconv(Lw[li][13], W1T, 256, 1024);
    u16* W2T  = wsub(1024, Nout); wconv(Lw[li][15], W2T, 1024, Nout);

    gemm(Ab, WqT, nullptr, nullptr, Qf, nullptr, M, 256, 128, 0, 0.17677669529663687f);
    gemm(Ab, WkT, nullptr, nullptr, Kf, nullptr, M, 256, 128, 0, 1.0f);
    gemm(Ab, WvT, nullptr, nullptr, Vf, nullptr, M, 256, 256, 0, 1.0f);
    gemm(Ab, Wg1T, nullptr, nullptr, nullptr, T1, M, 256, 16, 0, 1.0f);
    gemm(T1, Wg2T, Lw[li][5], nullptr, GKf, nullptr, M, 16, 128, 2, 1.0f);  // log_sigmoid/16
    gemm(Ab, WgT, nullptr, nullptr, Gf, nullptr, M, 256, 256, 0, 1.0f);

    scan_kernel<<<Bn * 4, 256, 0, stream>>>(Qf, Kf, Vf, GKf, Of, curT);
    gate_norm_kernel<<<M, 256, 0, stream>>>(Of, Gf, Lw[li][7], OB);

    gemm(OB, WoT, nullptr, nullptr, nullptr, ABf, M, 256, 256, 0, 1.0f);
    gemm(ABf, W0T, Lw[li][10], Af, X1, nullptr, M, 256, 256, 0, 1.0f);      // + residual
    layernorm_kernel<<<M, 256, 0, stream>>>(X1, Lw[li][11], Lw[li][12], XN);
    gemm(XN, W1T, Lw[li][14], nullptr, nullptr, H1, M, 256, 1024, 1, 1.0f); // GELU
    gemm(H1, W2T, Lw[li][16], (Nout == 256) ? Af : nullptr, Hf,
         (li == 7) ? Hb : nullptr, M, 1024, Nout, 0, 1.0f);

    if (li < 7) {
      dim3 g((unsigned)(Nout / 32), (unsigned)(curT / 32), (unsigned)Bn);
      transpose_kernel<<<g, dim3(32, 8), 0, stream>>>(Hf, Af, Ab, curT, Nout);
      curT = Nout;
    }
  }

  // 3) Final MLP head: 4096 -> 1024 -> 256 -> 64 -> 3 (GELU between)
  {
    size_t wo = 0;
    auto wsub = [&](int K, int N) -> u16* { u16* p = Wpool + wo; wo += (size_t)K * N; return p; };
    u16* F1T = wsub(4096, 1024); wconv(FW[0], F1T, 4096, 1024);
    u16* F2T = wsub(1024, 256);  wconv(FW[1], F2T, 1024, 256);
    u16* F3T = wsub(256, 64);    wconv(FW[2], F3T, 256, 64);
    u16* F4T = wsub(64, 3);      wconv(FW[3], F4T, 64, 3);
    gemm(Hb,  F1T, Fbias[0], nullptr, nullptr, FA,  256, 4096, 1024, 1, 1.0f);
    gemm(FA,  F2T, Fbias[1], nullptr, nullptr, FBx, 256, 1024, 256, 1, 1.0f);
    gemm(FBx, F3T, Fbias[2], nullptr, nullptr, FC,  256, 256, 64, 1, 1.0f);
    gemm(FC,  F4T, Fbias[3], nullptr, (float*)d_out, nullptr, 256, 64, 3, 0, 1.0f);
  }
}
